// SimpleLSTM_19688130085226
// MI455X (gfx1250) — compile-verified, matrix operand order checked by reading, not run on hardware
//
#include <hip/hip_runtime.h>

// ---------------------------------------------------------------------------
// Persistent-grid LSTM for MI455X (gfx1250, wave32, WMMA + TDM).
//
//  * prep kernel: W_w (2048x768 f32) -> bf16 WMMA-B fragments in ws.
//  * persistent kernel, 32 WGs x 256 threads (8 waves):
//      - WG wg owns hidden units h0=16*wg..h0+15 -> 4 WMMA n-tiles (F,I,O,G~).
//        96 KB of weight fragments pulled into LDS once via TENSOR_LOAD_TO_LDS
//        (TDM, TENSORcnt) and kept resident for all 2048 steps.
//      - Wave w computes batch rows 16w..16w+15; 24 kt x 4 nt = 96
//        v_wmma_f32_16x16x32_bf16 per wave per step (K = 768 over [H|Xe]).
//      - Activation buffer stored in WMMA A-FRAGMENT layout
//        [mtile][ktile][lane][32B] -> A loads are 1KB-contiguous, coalesced.
//      - Bias is folded by seeding the first WMMA's C operand from persistent
//        bias registers (no per-step accumulator-init movs).
//      - Cell state C lives in VGPRs (matches WMMA C/D lane layout).
//      - Xe[t+1] gathered from E and staged during step t (b64 stores).
//      - One grid-wide barrier per step: monotonic counter, relaxed polling,
//        single acquire fence on exit; counter zeroed by captured memset.
//
// Assumptions: ws_size >= ~3.5 MB; 32 WGs co-resident; 16-bit WMMA B lane
// layout mirrors the documented A layout (consistent convention between prep
// and compute).
// ---------------------------------------------------------------------------

#define VOCABSZ 100000
#define HIDDEN  512
#define EMBED   256
#define NBATCH  128
#define SEQLEN  2048
#define KTOT    768            // HIDDEN + EMBED
#define NKT     24             // KTOT / 32 k-tiles
#define NWG     32             // workgroups; each owns 16 hidden units
#define BLOCK   256            // 8 waves
#define NWAVE   8

#define A_OFF       256
#define ASTRIDE     (NBATCH * KTOT)          // ushorts per activation buffer
#define ABUF_BYTES  (2 * ASTRIDE * 2)        // double-buffered bf16
#define W_OFF       ((A_OFF + ABUF_BYTES + 1023) & ~1023)
#define FRAG_BYTES  1024                     // 32 lanes * 32B
#define FRAG_USH    512                      // ushorts per fragment
#define NTILES      128                      // 2048 gate cols / 16
#define LDS_BYTES   (4 * NKT * FRAG_BYTES)   // 96 KB per WG
#define CHUNK_B     (NKT * FRAG_BYTES)       // 24 KB per gate slice
#define CHUNK_QW    (CHUNK_B / 8)            // 3072 8-byte elements

typedef __attribute__((ext_vector_type(4)))  unsigned int u32x4;
typedef __attribute__((ext_vector_type(2)))  unsigned int u32x2;
typedef __attribute__((ext_vector_type(4)))  int          i32x4;
typedef __attribute__((ext_vector_type(8)))  int          i32x8;
typedef __attribute__((ext_vector_type(4)))  float        f32x4;
typedef __attribute__((ext_vector_type(8)))  float        v8f;
typedef __attribute__((ext_vector_type(16))) __bf16       v16bf;

#if defined(__has_builtin)
#  if __has_builtin(__builtin_amdgcn_tensor_load_to_lds) && \
      __has_builtin(__builtin_amdgcn_s_wait_tensorcnt)
#    define USE_TDM 1
#  endif
#endif
#ifndef USE_TDM
#  define USE_TDM 0
#endif

union AFrag { u32x4 u[2]; v16bf v; };

__device__ __forceinline__ unsigned short bf16b(float x) {
    unsigned u = __float_as_uint(x);
    u += 0x7FFFu + ((u >> 16) & 1u);         // round-to-nearest-even
    return (unsigned short)(u >> 16);
}
__device__ __forceinline__ unsigned bfpack(float lo, float hi) {
    return (unsigned)bf16b(lo) | ((unsigned)bf16b(hi) << 16);
}
__device__ __forceinline__ float sigm_f(float x) {
    float e = __builtin_amdgcn_exp2f(-1.4426950408889634f * x);
    return __builtin_amdgcn_rcpf(1.0f + e);
}
__device__ __forceinline__ float tanh_f(float x) {
    float e = __builtin_amdgcn_exp2f(2.8853900817779268f * x);   // 2^(2x*log2e)
    return 1.0f - 2.0f * __builtin_amdgcn_rcpf(e + 1.0f);
}

// Grid-wide barrier: monotonic counter. Relaxed polling (no per-poll cache
// maintenance), one agent-scope acquire fence once the round completes.
__device__ __forceinline__ void grid_barrier(unsigned* cnt, unsigned target) {
    __syncthreads();
    if (threadIdx.x == 0) {
        __hip_atomic_fetch_add(cnt, 1u, __ATOMIC_RELEASE, __HIP_MEMORY_SCOPE_AGENT);
        while (__hip_atomic_load(cnt, __ATOMIC_RELAXED, __HIP_MEMORY_SCOPE_AGENT) < target)
            __builtin_amdgcn_s_sleep(1);
        __builtin_amdgcn_fence(__ATOMIC_ACQUIRE, "agent");
    }
    __syncthreads();
}

// Fragment-layout element mapping for (row lm = m&15, kk = k&31):
//   lane  = lm + (kk&8 ? 16 : 0)
//   dword = ((kk>>4)<<2) + ((kk&7)>>1),  half = kk&1
// Fragment index = (m>>4)*NKT + (k>>5), 1024B each.

// Stage Xe[t_next] (bf16) for this WG's 4 batch rows into next activation buf.
__device__ __forceinline__ void stage_xe(int tn, const int* __restrict__ X,
                                         const float* __restrict__ E,
                                         unsigned short* __restrict__ Anxt,
                                         int wg, int tid) {
    int m = wg * 4 + (tid >> 6);             // 4 rows per WG, 64 threads/row
    int e = (tid & 63) << 2;                 // 4 embed elems per thread
    int tok = X[m * SEQLEN + tn];
    f32x4 ev = *(const f32x4*)(E + (long)tok * EMBED + e);
    u32x2 v; v.x = bfpack(ev.x, ev.y); v.y = bfpack(ev.z, ev.w);
    int kk    = e & 31;
    int ktx   = 16 + (e >> 5);               // k = 512 + e
    int lane2 = (m & 15) + ((kk & 8) ? 16 : 0);
    int dw    = ((kk >> 4) << 2) + ((kk & 7) >> 1);   // even -> 8B aligned
    unsigned short* dst = Anxt + ((m >> 4) * NKT + ktx) * FRAG_USH
                               + lane2 * 16 + dw * 2;
    *(u32x2*)dst = v;
}

// --- prep: f32 W (row-major [2048][768]) -> bf16 B fragments [nt][kt][lane] --
__global__ void prep_weights(const float* __restrict__ W, u32x4* __restrict__ wsW) {
    int wave = blockIdx.x * NWAVE + (threadIdx.x >> 5);
    int lane = threadIdx.x & 31;
    int nt = wave / NKT, kt = wave - nt * NKT;
    int n  = nt * 16 + (lane & 15);
    int k0 = kt * 32 + ((lane >> 4) << 3);   // lanes 16-31: K offset +8
    const float* wr = W + n * KTOT + k0;
    f32x4 a0 = *(const f32x4*)(wr + 0);
    f32x4 a1 = *(const f32x4*)(wr + 4);
    f32x4 b0 = *(const f32x4*)(wr + 16);
    f32x4 b1 = *(const f32x4*)(wr + 20);
    u32x4 lo = { bfpack(a0.x, a0.y), bfpack(a0.z, a0.w),
                 bfpack(a1.x, a1.y), bfpack(a1.z, a1.w) };
    u32x4 hi = { bfpack(b0.x, b0.y), bfpack(b0.z, b0.w),
                 bfpack(b1.x, b1.y), bfpack(b1.z, b1.w) };
    u32x4* dst = wsW + (nt * NKT + kt) * 64 + lane * 2;
    dst[0] = lo;
    dst[1] = hi;
}

// ------------------------------ main kernel --------------------------------
__global__ void __launch_bounds__(BLOCK)
lstm_persistent(const int* __restrict__ X, const float* __restrict__ E,
                const float* __restrict__ Wb, const u32x4* __restrict__ wsW,
                unsigned short* __restrict__ Abuf, unsigned* __restrict__ cnt,
                float* __restrict__ out) {
    extern __shared__ char smem[];
    const int tid  = threadIdx.x;
    const int wg   = blockIdx.x;
    const int wave = tid >> 5;
    const int lane = tid & 31;
    const int h0   = wg * 16;
    const int m0   = wave * 16;

    // --- preload this WG's 96 KB of weight fragments into LDS -------------
#if USE_TDM
    if (tid < 32) {                          // wave 0 issues 4 TDM descriptors
        unsigned lds_base = (unsigned)(unsigned long long)(&smem[0]);
        for (int g = 0; g < 4; ++g) {
            unsigned long long ga = (unsigned long long)
                (const void*)(wsW + ((wg + 32 * g) * NKT) * 64);
            u32x4 d0;
            d0.x = 1u;                                    // count=1, user load
            d0.y = lds_base + (unsigned)(g * CHUNK_B);    // lds_addr
            d0.z = (unsigned)(ga & 0xFFFFFFFFu);          // global_addr lo
            d0.w = (unsigned)((ga >> 32) & 0x01FFFFFFu)   // global_addr hi
                 | 0x80000000u;                           // type = 2 (image)
            i32x8 d1;
            d1[0] = 0x00030000;                 // data_size = 8B, mask=0
            d1[1] = (int)(CHUNK_QW << 16);      // tensor_dim0 (lo16 @ b63:48)
            d1[2] = (1 << 16);                  // tensor_dim1 = 1
            d1[3] = (int)(CHUNK_QW << 16);      // tile_dim0 @ b127:112
            d1[4] = 1;                          // tile_dim1 = 1
            d1[5] = CHUNK_QW;                   // tensor_dim0_stride
            d1[6] = 0;
            d1[7] = 0;
            i32x4 z4 = {0, 0, 0, 0};
#if __clang_major__ >= 23
            i32x8 z8 = {0, 0, 0, 0, 0, 0, 0, 0};
            __builtin_amdgcn_tensor_load_to_lds(d0, d1, z4, z4, z8, 0);
#else
            __builtin_amdgcn_tensor_load_to_lds(d0, d1, z4, z4, 0);
#endif
        }
        __builtin_amdgcn_s_wait_tensorcnt(0);
    }
#else
    {
        u32x4* lds4 = (u32x4*)smem;
        for (int i = tid; i < (LDS_BYTES / 16); i += BLOCK) {
            int g   = i / (NKT * 64);
            int rem = i - g * (NKT * 64);
            lds4[i] = wsW[((wg + 32 * g) * NKT) * 64 + rem];
        }
    }
#endif

    // Persistent bias seed: C operand of the first WMMA of every step.
    v8f bseed[4];
#pragma unroll
    for (int g = 0; g < 4; ++g) {
        float bv = Wb[g * HIDDEN + h0 + (lane & 15)];
#pragma unroll
        for (int r = 0; r < 8; ++r) bseed[g][r] = bv;
    }

    v8f C;
#pragma unroll
    for (int r = 0; r < 8; ++r) C[r] = 0.0f;

    __syncthreads();

    // Stage Xe[0] into buffer 0 (H fragments of both buffers pre-zeroed)
    stage_xe(0, X, E, Abuf, wg, tid);
    grid_barrier(cnt, NWG);

    const int rowm   = (lane & 15);
    const int kshift = (lane >> 4) << 3;     // lanes 16-31 take K+8 halves

    // H-store addressing (fragment layout), constant across steps:
    // this lane writes hidden column k = h0 + rowm for 8 rows.
    const int kH   = h0 + rowm;
    const int kkH  = kH & 31;
    const int hOff = (kH >> 5) * FRAG_USH                 // ktile
                   + ((kkH & 8) ? 16 : 0) * 16            // lane-half select
                   + (((kkH >> 4) << 2) + ((kkH & 7) >> 1)) * 2
                   + (kkH & 1);

    for (int t = 0; t < SEQLEN; ++t) {
        const unsigned short* Acur = Abuf + (t & 1) * ASTRIDE;
        unsigned short*       Anxt = Abuf + ((t + 1) & 1) * ASTRIDE;

        // Coalesced A-fragment base: this wave's m-tile, per-lane 32B block.
        const unsigned short* afb = Acur + (wave * NKT) * FRAG_USH + lane * 16;

        v8f acc[4];
        {   // kt = 0: seed accumulators from persistent bias registers (C!=D)
            AFrag af;
            const u32x4* p = (const u32x4*)afb;
            af.u[0] = p[0];
            af.u[1] = p[1];
#pragma unroll
            for (int g = 0; g < 4; ++g) {
                v16bf b = *(const v16bf*)(smem + ((g * NKT) << 10) + (lane << 5));
                acc[g] = __builtin_amdgcn_wmma_f32_16x16x32_bf16(
                    false, af.v, false, b, (short)0, bseed[g], false, false);
            }
        }
#pragma unroll
        for (int kt = 1; kt < NKT; ++kt) {
            AFrag af;
            const u32x4* p = (const u32x4*)(afb + kt * FRAG_USH);
            af.u[0] = p[0];
            af.u[1] = p[1];
#pragma unroll
            for (int g = 0; g < 4; ++g) {
                v16bf b = *(const v16bf*)(smem + ((g * NKT + kt) << 10) + (lane << 5));
                acc[g] = __builtin_amdgcn_wmma_f32_16x16x32_bf16(
                    false, af.v, false, b, (short)0, acc[g], false, false);
            }
        }

        // Stage Xe for t+1 while WMMA results settle
        if (t + 1 < SEQLEN) stage_xe(t + 1, X, E, Anxt, wg, tid);

        // Gates + state update; C/D layout: VGPR r, lane l ->
        //   row = m0 + r + 8*(l>=16), col = h0 + (l&15)
        unsigned short* hdst = Anxt + (wave * NKT) * FRAG_USH + hOff;
        const int row = m0 + kshift;
#pragma unroll
        for (int r = 0; r < 8; ++r) {
            float f  = sigm_f(acc[0][r]);
            float i  = sigm_f(acc[1][r]);
            float o  = sigm_f(acc[2][r]);
            float gg = tanh_f(acc[3][r]);
            float c  = f * C[r] + i * gg;
            C[r] = c;
            float h = o * tanh_f(c);
            hdst[(kshift + r) * 16] = bf16b(h);      // fragment-layout H write
            if (t == SEQLEN - 1)
                out[(row + r) * HIDDEN + h0 + rowm] = h;
        }

        grid_barrier(cnt, (unsigned)(t + 2) * NWG);
    }
}

// ------------------------------ launcher -----------------------------------
extern "C" void kernel_launch(void* const* d_in, const int* in_sizes, int n_in,
                              void* d_out, int out_size, void* d_ws, size_t ws_size,
                              hipStream_t stream) {
    (void)in_sizes; (void)n_in; (void)out_size; (void)ws_size;
    const int*   X  = (const int*)d_in[0];
    const float* E  = (const float*)d_in[1];
    const float* Ww = (const float*)d_in[2];
    const float* Wb = (const float*)d_in[3];
    float*       out = (float*)d_out;

    unsigned*       cnt  = (unsigned*)d_ws;
    unsigned short* Abuf = (unsigned short*)((char*)d_ws + A_OFF);
    u32x4*          wsW  = (u32x4*)((char*)d_ws + W_OFF);

    // Zero barrier counter + both activation buffers (H0 = 0). Captured as a
    // graph node, so every replay is deterministic.
    hipMemsetAsync(d_ws, 0, A_OFF + ABUF_BYTES, stream);

    prep_weights<<<(NTILES * NKT) / NWAVE, BLOCK, 0, stream>>>(Ww, wsW);

    hipFuncSetAttribute((const void*)lstm_persistent,
                        hipFuncAttributeMaxDynamicSharedMemorySize, LDS_BYTES);
    lstm_persistent<<<NWG, BLOCK, LDS_BYTES, stream>>>(X, E, Wb, wsW, Abuf, cnt, out);
}
